// CGCNN_57698590654845
// MI455X (gfx1250) — matrix operand order repaired and checked by reading, not compiled, not run on hardware
//
#include <hip/hip_runtime.h>
#include <hip/hip_bf16.h>

// ---------------------------------------------------------------------------
// CGCNN on MI455X (gfx1250, wave32, WMMA).
//
// Decomposition: z@W = h[dst]@W_dst + h[src]@W_src + e@W_e.
//  - Node products P16 = h16 @ [Wf_dst|Wf_src|Ws_dst|Ws_src]  (WMMA GEMM,
//    M=100000=6250*16, N=512, K=128). Wcat16 is stored PRE-SWIZZLED into the
//    WMMA B-fragment layout so each lane's B fragment is one 32B load.
//  - Edge kernel: per wave a 16-edge tile; edge_attr (16x32) is one WMMA
//    A-fragment (two b128 LDS loads per lane) against W_e (32x128 f16 staged
//    swizzled in LDS -> one 32B ds read per fragment) ->
//    16 v_wmma_f32_16x16x32_f16 per wave; node terms gathered from P16;
//    sigmoid*softplus; scatter via global_atomic_add_f32.
// Memory-bound by design (~8.4 GB edge traffic -> ~360us @ 23.3 TB/s);
// node arrays (~100MB) largely resident in 192MB L2 so gathers hit L2.
//
// Workspace (assumes ws_size >= ~232 MB):
//   h:12.8M f32  agg:12.8M f32  h16:12.8M f16  P16:51.2M f16
//   Wcat16(swizzled):65536 f16  agg1:300K f32  sums:8192 f32 + cnt:64 f32
// ---------------------------------------------------------------------------

#define N_NODES   100000
#define N_EDGES   1600000
#define HIDDEN    128
#define N_GRAPHS  64

typedef _Float16 h4v  __attribute__((ext_vector_type(4)));
typedef _Float16 h8v  __attribute__((ext_vector_type(8)));
typedef _Float16 h16v __attribute__((ext_vector_type(16)));
typedef float    f8v  __attribute__((ext_vector_type(8)));

// ---------------- generic zero ----------------
__global__ void k_zero(float* __restrict__ p, long long n) {
  long long i = (long long)blockIdx.x * blockDim.x + threadIdx.x;
  long long stride = (long long)gridDim.x * blockDim.x;
  for (; i < n; i += stride) p[i] = 0.0f;
}

// ---------------- conv1 (channels=3, scalar) ----------------
__global__ void k_conv1_edge(const float* __restrict__ x, const int* __restrict__ ei,
                             const float* __restrict__ ea,
                             const float* __restrict__ Wf, const float* __restrict__ bf,
                             const float* __restrict__ Ws, const float* __restrict__ bs,
                             float* __restrict__ agg1) {
  long long e = (long long)blockIdx.x * blockDim.x + threadIdx.x;
  if (e >= N_EDGES) return;
  const int s = ei[e];
  const int d = ei[N_EDGES + e];
  float accf[3], accs[3];
  #pragma unroll
  for (int o = 0; o < 3; ++o) { accf[o] = bf[o]; accs[o] = bs[o]; }
  #pragma unroll
  for (int c = 0; c < 3; ++c) {
    const float xd = x[(long long)d * 3 + c];
    const float xs = x[(long long)s * 3 + c];
    #pragma unroll
    for (int o = 0; o < 3; ++o) {
      accf[o] += xd * Wf[c * 3 + o] + xs * Wf[(3 + c) * 3 + o];
      accs[o] += xd * Ws[c * 3 + o] + xs * Ws[(3 + c) * 3 + o];
    }
  }
  const float4* ear = (const float4*)(ea + e * 32);   // 128B-aligned rows
  #pragma unroll
  for (int kq = 0; kq < 8; ++kq) {
    const float4 ev = ear[kq];
    const float evv[4] = {ev.x, ev.y, ev.z, ev.w};
    #pragma unroll
    for (int q = 0; q < 4; ++q) {
      const int k = kq * 4 + q;
      #pragma unroll
      for (int o = 0; o < 3; ++o) {
        accf[o] += evv[q] * Wf[(6 + k) * 3 + o];
        accs[o] += evv[q] * Ws[(6 + k) * 3 + o];
      }
    }
  }
  #pragma unroll
  for (int o = 0; o < 3; ++o) {
    const float sig = 1.0f / (1.0f + __expf(-accf[o]));
    const float sp  = (accs[o] > 20.0f) ? accs[o] : log1pf(__expf(accs[o]));
    atomicAdd(&agg1[(long long)d * 3 + o], sig * sp);
  }
}

// ---------------- projection: h = relu((x+agg1) @ Wp + bp) ----------------
__global__ void k_proj(const float* __restrict__ x, const float* __restrict__ agg1,
                       const float* __restrict__ Wp, const float* __restrict__ bp,
                       float* __restrict__ h, _Float16* __restrict__ h16) {
  const int n = blockIdx.x;
  const int j = threadIdx.x;      // 0..127
  float acc = bp[j];
  #pragma unroll
  for (int c = 0; c < 3; ++c)
    acc += (x[(long long)n * 3 + c] + agg1[(long long)n * 3 + c]) * Wp[c * HIDDEN + j];
  acc = fmaxf(acc, 0.0f);
  h  [(long long)n * HIDDEN + j] = acc;
  h16[(long long)n * HIDDEN + j] = (_Float16)acc;
}

// ---- pack Wcat16 into WMMA B-fragment-major layout ------------------------
// Logical Wcat (128x512): cols [0:128)=Wf_dst, [128:256)=Wf_src,
// [256:384)=Ws_dst, [384:512)=Ws_src.
// Swizzled: Wsw[(((kc*32)+nt)*32 + lane)*16 + j] = Wcat[kc*32 + (j>>3)*16 + (lane>>4)*8 + (j&7)]
//                                                      [nt*16 + (lane&15)]
__global__ void k_build_wcat(const float* __restrict__ Wf, const float* __restrict__ Ws,
                             _Float16* __restrict__ Wsw) {
  const int i = blockIdx.x * blockDim.x + threadIdx.x;   // 0..65535
  if (i >= 128 * 512) return;
  const int j  = i & 15;
  const int ln = (i >> 4) & 31;
  const int nt = (i >> 9) & 31;
  const int kc = i >> 14;                                // 0..3
  const int k = kc * 32 + ((j >> 3) << 4) + ((ln >> 4) << 3) + (j & 7);  // logical K row 0..127
  const int c = nt * 16 + (ln & 15);                     // logical col 0..511
  float v;
  if      (c < 128) v = Wf[k * HIDDEN + c];
  else if (c < 256) v = Wf[(128 + k) * HIDDEN + (c - 128)];
  else if (c < 384) v = Ws[k * HIDDEN + (c - 256)];
  else              v = Ws[(128 + k) * HIDDEN + (c - 384)];
  Wsw[i] = (_Float16)v;
}

// ---------------- WMMA node GEMM: P16[M,512] = h16[M,128] @ Wcat ----------
// One wave per block; each wave computes 4 N-tiles (A fragment reused 4x).
__global__ void k_node_gemm(const _Float16* __restrict__ A,   // M x 128
                            const _Float16* __restrict__ Bsw, // swizzled 128x512
                            _Float16* __restrict__ P) {       // M x 512
  const int lane = threadIdx.x;          // blockDim.x == 32
  const int mt = blockIdx.x;             // 0..6249
  const int ng = blockIdx.y;             // 0..7 (group of 4 N-tiles)
  const int mrow = lane & 15;
  const int hi = lane >> 4;
  const _Float16* arow = A + ((long long)mt * 16 + mrow) * HIDDEN + (hi ? 8 : 0);
  const h16v* bw = (const h16v*)Bsw;     // 32B-aligned fragments
  f8v c0 = {}, c1 = {}, c2 = {}, c3 = {};
  #pragma unroll
  for (int kc = 0; kc < 4; ++kc) {
    const int kb = kc * 32;
    // A fragment: two contiguous 8-half runs (16B each) per lane.
    const h8v a0 = *(const h8v*)(arow + kb);
    const h8v a1 = *(const h8v*)(arow + kb + 16);
    const h16v a = __builtin_shufflevector(a0, a1, 0,1,2,3,4,5,6,7,8,9,10,11,12,13,14,15);
    const int fb = (kc * 32 + ng * 4) * 32 + lane;
    const h16v b0 = bw[fb];
    const h16v b1 = bw[fb + 32];
    const h16v b2 = bw[fb + 64];
    const h16v b3 = bw[fb + 96];
    c0 = __builtin_amdgcn_wmma_f32_16x16x32_f16(false, a, false, b0, (short)0, c0, false, false);
    c1 = __builtin_amdgcn_wmma_f32_16x16x32_f16(false, a, false, b1, (short)0, c1, false, false);
    c2 = __builtin_amdgcn_wmma_f32_16x16x32_f16(false, a, false, b2, (short)0, c2, false, false);
    c3 = __builtin_amdgcn_wmma_f32_16x16x32_f16(false, a, false, b3, (short)0, c3, false, false);
  }
  const f8v acc[4] = {c0, c1, c2, c3};
  #pragma unroll
  for (int t = 0; t < 4; ++t) {
    const int ncol = (ng * 4 + t) * 16 + (lane & 15);
    #pragma unroll
    for (int r = 0; r < 8; ++r) {
      const int m = r + (hi ? 8 : 0);  // C/D: lanes<16 -> M=r, lanes>=16 -> M=8+r
      P[((long long)mt * 16 + m) * 512 + ncol] = (_Float16)acc[t][r];
    }
  }
}

// ---------------- fused edge kernel (WMMA on edge_attr @ W_e) ----------------
// 256 threads = 8 waves; each wave handles a tile of 16 edges.
__global__ void k_edge(const float* __restrict__ ea, const int* __restrict__ ei,
                       const _Float16* __restrict__ P16,
                       const float* __restrict__ Wf, const float* __restrict__ Ws,
                       const float* __restrict__ bf, const float* __restrict__ bs,
                       float* __restrict__ agg) {
  // W_e staged swizzled: fragment (nt,lane) at sWe[(nt*32+lane)*16], nt 0..7=f,
  // 8..15=s; element j = logical We[(j>>3)*16 + (lane>>4)*8 + (j&7)][nt*16 + (lane&15)].
  __shared__ __align__(32) _Float16 sWe[16 * 32 * 16];    // 16KB
  __shared__ __align__(32) _Float16 sEA[8][16 * 32];      // per-wave 16x32 tile
  __shared__ int sSrc[8][16];
  __shared__ int sDst[8][16];

  const int tid  = threadIdx.x;
  const int wv   = tid >> 5;
  const int lane = tid & 31;

  for (int i = tid; i < 16 * 32 * 16; i += 256) {
    const int j  = i & 15;
    const int ln = (i >> 4) & 31;
    const int nt = i >> 9;                       // 0..15
    const int k = ((j >> 3) << 4) + ((ln >> 4) << 3) + (j & 7);  // 0..31
    const int c = nt * 16 + (ln & 15);           // 0..255
    const float v = (c < 128) ? Wf[(256 + k) * HIDDEN + c]
                              : Ws[(256 + k) * HIDDEN + (c - 128)];
    sWe[i] = (_Float16)v;
  }
  const long long ebase = ((long long)blockIdx.x * 8 + wv) * 16;
  // Stage edge_attr tile: coalesced float4 reads, 8B f16 LDS writes.
  const float4* ear = (const float4*)(ea + ebase * 32);
  h4v* sEA4 = (h4v*)&sEA[wv][0];
  #pragma unroll
  for (int j = 0; j < 4; ++j) {
    const int flat = j * 128 + lane * 4;         // 0..508
    const float4 v = ear[flat >> 2];
    h4v hv;
    hv[0] = (_Float16)v.x; hv[1] = (_Float16)v.y;
    hv[2] = (_Float16)v.z; hv[3] = (_Float16)v.w;
    sEA4[flat >> 2] = hv;
  }
  if (lane < 16) {
    sSrc[wv][lane] = ei[ebase + lane];
    sDst[wv][lane] = ei[(long long)N_EDGES + ebase + lane];
  }
  __syncthreads();

  // A fragment: 16 edges x 32 features; two contiguous b128 LDS reads.
  const int mrow = lane & 15;
  const int hi = lane >> 4;
  const h8v* arow = (const h8v*)&sEA[wv][mrow * 32 + (hi ? 8 : 0)];
  const h8v a0 = arow[0];
  const h8v a1 = arow[2];                         // +16 halfs
  const h16v afrag = __builtin_shufflevector(a0, a1, 0,1,2,3,4,5,6,7,8,9,10,11,12,13,14,15);
  const h16v* swv = (const h16v*)sWe;

  for (int nt = 0; nt < 8; ++nt) {
    const h16v bF = swv[nt * 32 + lane];
    const h16v bS = swv[(nt + 8) * 32 + lane];
    f8v cf = {}, cs = {};
    cf = __builtin_amdgcn_wmma_f32_16x16x32_f16(false, afrag, false, bF, (short)0, cf, false, false);
    cs = __builtin_amdgcn_wmma_f32_16x16x32_f16(false, afrag, false, bS, (short)0, cs, false, false);

    const int n = nt * 16 + (lane & 15);
    const float bfv = bf[n], bsv = bs[n];
    #pragma unroll
    for (int r = 0; r < 8; ++r) {
      const int m = r + (hi ? 8 : 0);
      const int d = sDst[wv][m];
      const int s = sSrc[wv][m];
      const long long pd = (long long)d * 512;
      const long long ps = (long long)s * 512;
      const float gf = cf[r] + (float)P16[pd + n]       + (float)P16[ps + 128 + n] + bfv;
      const float gs = cs[r] + (float)P16[pd + 256 + n] + (float)P16[ps + 384 + n] + bsv;
      const float sig = 1.0f / (1.0f + __expf(-gf));
      const float sp  = (gs > 20.0f) ? gs : log1pf(__expf(gs));
      atomicAdd(&agg[(long long)d * HIDDEN + n], sig * sp);
    }
  }
}

// ---------------- h = relu(h + agg); refresh h16 ----------------
__global__ void k_update(float* __restrict__ h, const float* __restrict__ agg,
                         _Float16* __restrict__ h16, long long n) {
  long long i = (long long)blockIdx.x * blockDim.x + threadIdx.x;
  const long long stride = (long long)gridDim.x * blockDim.x;
  for (; i < n; i += stride) {
    const float v = fmaxf(h[i] + agg[i], 0.0f);
    h[i] = v;
    h16[i] = (_Float16)v;
  }
}

// ---------------- mean pool (atomics) ----------------
__global__ void k_pool(const float* __restrict__ h, const int* __restrict__ batch,
                       float* __restrict__ sums, float* __restrict__ cnt) {
  const int n = blockIdx.x;
  const int j = threadIdx.x;
  const int g = batch[n];
  atomicAdd(&sums[g * HIDDEN + j], h[(long long)n * HIDDEN + j]);
  if (j == 0) atomicAdd(&cnt[g], 1.0f);
}

// ---------------- head: out = relu(pooled@W1+b1) @ W2 + b2 ----------------
__global__ void k_head(const float* __restrict__ sums, const float* __restrict__ cnt,
                       const float* __restrict__ W1, const float* __restrict__ b1,
                       const float* __restrict__ W2, const float* __restrict__ b2,
                       float* __restrict__ out) {
  __shared__ float pooled[HIDDEN];
  __shared__ float gv[HIDDEN];
  const int g = blockIdx.x;
  const int j = threadIdx.x;
  const float c = fmaxf(cnt[g], 1.0f);
  pooled[j] = sums[g * HIDDEN + j] / c;
  __syncthreads();
  float acc = b1[j];
  for (int k = 0; k < HIDDEN; ++k) acc += pooled[k] * W1[k * HIDDEN + j];
  gv[j] = fmaxf(acc, 0.0f);
  __syncthreads();
  if (j < 3) {
    float acc2 = b2[j];
    for (int k = 0; k < HIDDEN; ++k) acc2 += gv[k] * W2[k * 3 + j];
    out[g * 3 + j] = acc2;
  }
}

// ---------------------------------------------------------------------------
extern "C" void kernel_launch(void* const* d_in, const int* in_sizes, int n_in,
                              void* d_out, int out_size, void* d_ws, size_t ws_size,
                              hipStream_t stream) {
  (void)in_sizes; (void)n_in; (void)out_size; (void)ws_size;
  const float* x   = (const float*)d_in[0];
  const int*   ei  = (const int*)  d_in[1];
  const float* ea  = (const float*)d_in[2];
  const int*   bat = (const int*)  d_in[3];
  const float* Wf1 = (const float*)d_in[4];
  const float* bf1 = (const float*)d_in[5];
  const float* Ws1 = (const float*)d_in[6];
  const float* bs1 = (const float*)d_in[7];
  const float* Wp  = (const float*)d_in[8];
  const float* bp  = (const float*)d_in[9];
  const float* Wf2 = (const float*)d_in[10];
  const float* bf2 = (const float*)d_in[11];
  const float* Ws2 = (const float*)d_in[12];
  const float* bs2 = (const float*)d_in[13];
  const float* Wf3 = (const float*)d_in[14];
  const float* bf3 = (const float*)d_in[15];
  const float* Ws3 = (const float*)d_in[16];
  const float* bs3 = (const float*)d_in[17];
  const float* W1  = (const float*)d_in[18];
  const float* b1  = (const float*)d_in[19];
  const float* W2  = (const float*)d_in[20];
  const float* b2  = (const float*)d_in[21];
  float* out = (float*)d_out;

  // Workspace carve-up (256B aligned).
  char* p = (char*)d_ws;
  auto bump = [&](size_t bytes) -> char* {
    char* r = p;
    p += (bytes + 255) & ~(size_t)255;
    return r;
  };
  const long long HN = (long long)N_NODES * HIDDEN;          // 12.8M
  float*    h      = (float*)   bump(HN * sizeof(float));
  float*    agg    = (float*)   bump(HN * sizeof(float));
  _Float16* h16    = (_Float16*)bump(HN * sizeof(_Float16));
  _Float16* P16    = (_Float16*)bump((long long)N_NODES * 512 * sizeof(_Float16));
  _Float16* Wcat16 = (_Float16*)bump(128 * 512 * sizeof(_Float16));
  float*    agg1   = (float*)   bump((long long)N_NODES * 3 * sizeof(float));
  float*    sums   = (float*)   bump((N_GRAPHS * HIDDEN + N_GRAPHS) * sizeof(float));
  float*    cnt    = sums + N_GRAPHS * HIDDEN;               // zeroed with sums

  const int ZB = 2048;  // blocks for grid-stride zero/update

  // ---- conv1 (tiny dims) + projection ----
  k_zero<<<ZB, 256, 0, stream>>>(agg1, (long long)N_NODES * 3);
  k_conv1_edge<<<(N_EDGES + 255) / 256, 256, 0, stream>>>(x, ei, ea, Wf1, bf1, Ws1, bs1, agg1);
  k_proj<<<N_NODES, HIDDEN, 0, stream>>>(x, agg1, Wp, bp, h, h16);

  // ---- conv2 ----
  k_build_wcat<<<(128 * 512) / 256, 256, 0, stream>>>(Wf2, Ws2, Wcat16);
  k_node_gemm<<<dim3(N_NODES / 16, 8), 32, 0, stream>>>(h16, Wcat16, P16);
  k_zero<<<ZB, 256, 0, stream>>>(agg, HN);
  k_edge<<<N_EDGES / 128, 256, 0, stream>>>(ea, ei, P16, Wf2, Ws2, bf2, bs2, agg);
  k_update<<<ZB, 256, 0, stream>>>(h, agg, h16, HN);

  // ---- conv3 ----
  k_build_wcat<<<(128 * 512) / 256, 256, 0, stream>>>(Wf3, Ws3, Wcat16);
  k_node_gemm<<<dim3(N_NODES / 16, 8), 32, 0, stream>>>(h16, Wcat16, P16);
  k_zero<<<ZB, 256, 0, stream>>>(agg, HN);
  k_edge<<<N_EDGES / 128, 256, 0, stream>>>(ea, ei, P16, Wf3, Ws3, bf3, bs3, agg);
  k_update<<<ZB, 256, 0, stream>>>(h, agg, h16, HN);

  // ---- pool + head ----
  k_zero<<<64, 256, 0, stream>>>(sums, N_GRAPHS * HIDDEN + N_GRAPHS);
  k_pool<<<N_NODES, HIDDEN, 0, stream>>>(h, bat, sums, cnt);
  k_head<<<N_GRAPHS, HIDDEN, 0, stream>>>(sums, cnt, W1, b1, W2, b2, out);
}
